// BiLSTM_CRF_45552423141694
// MI455X (gfx1250) — matrix operand order repaired and checked by reading, not compiled, not run on hardware
//
#include <hip/hip_runtime.h>
#include <hip/hip_bf16.h>

// ---------------------------------------------------------------------------
// Problem constants (BiLSTM-CRF): B=64, T=512, V=8000, E=128, D=512, K=35, H=256
// ---------------------------------------------------------------------------
#define BB   64
#define TT   512
#define EE   128
#define DD   512
#define HH   256
#define GG   1024      // 4*H gate width
#define KK   35
#define NPAD 48        // K padded to 3 WMMA n-tiles
#define KSTART 33
#define KSTOP  34
#define MROWS (BB*TT)  // 32768

typedef __attribute__((ext_vector_type(16))) __bf16 v16bf;
typedef __attribute__((ext_vector_type(8)))  float  v8f;

struct FragBF { union { v16bf v; uint4 q[2]; }; };

__device__ __forceinline__ unsigned short f32_to_bf16u(float f) {
    union { float f; unsigned u; } v; v.f = f;
    unsigned r = v.u + 0x7FFFu + ((v.u >> 16) & 1u);   // RNE
    return (unsigned short)(r >> 16);
}

__device__ __forceinline__ float sigf(float x) { return 1.0f / (1.0f + __expf(-x)); }

// A-fragment (16x32 bf16, M x K). Row-major source, row stride ld (elements).
// ISA layout: lanes 0-15 row M=lane, K 0..7 (q0) and 16..23 (q1);
//             lanes 16-31 row M=lane-16, K 8..15 and 24..31.
__device__ __forceinline__ void load_a_frag(FragBF& f, const unsigned short* base,
                                            int row0, int k0, int ld, int lane) {
    int r = row0 + (lane & 15);
    int kofs = (lane >= 16) ? 8 : 0;
    const unsigned short* p = base + (size_t)r * ld + k0 + kofs;
    f.q[0] = *(const uint4*)(p);
    f.q[1] = *(const uint4*)(p + 16);
}

// B-fragment (32x16 bf16, K x N): source holds W[n][k] row-major (computes X@W.T).
// lane%16 = column n; lanes 0-15 hold K 0..7 and 16..23, lanes 16-31 hold K 8..15 and 24..31.
__device__ __forceinline__ void load_b_frag(FragBF& f, const unsigned short* base,
                                            int col0, int k0, int ld, int lane) {
    int c = col0 + (lane & 15);
    int kofs = (lane >= 16) ? 8 : 0;
    const unsigned short* p = base + (size_t)c * ld + k0 + kofs;
    f.q[0] = *(const uint4*)(p);
    f.q[1] = *(const uint4*)(p + 16);
}

__device__ __forceinline__ v8f wmma_bf16(const FragBF& a, const FragBF& b, v8f c) {
    return __builtin_amdgcn_wmma_f32_16x16x32_bf16(false, a.v, false, b.v,
                                                   (short)0, c, false, false);
}

// ---------------------------------------------------------------------------
// Utility kernels
// ---------------------------------------------------------------------------
__global__ void cvt_f32_bf16_kernel(const float* __restrict__ src,
                                    unsigned short* __restrict__ dst, int n) {
    int i = blockIdx.x * 256 + threadIdx.x;
    if (i < n) dst[i] = f32_to_bf16u(src[i]);
}

// Wout (35x512) -> zero-padded 48x512 bf16
__global__ void pad_wout_kernel(const float* __restrict__ Wout,
                                unsigned short* __restrict__ dst) {
    int i = blockIdx.x * 256 + threadIdx.x;
    if (i >= NPAD * DD) return;
    int n = i / DD;
    dst[i] = (n < KK) ? f32_to_bf16u(Wout[i]) : (unsigned short)0;
}

__global__ void embed_kernel(const int* __restrict__ sent,
                             const float* __restrict__ embed,
                             unsigned short* __restrict__ xbf) {
    int i = blockIdx.x * 256 + threadIdx.x;     // over B*T*E
    if (i >= MROWS * EE) return;
    int bt = i >> 7;            // /128
    int e  = i & 127;
    xbf[i] = f32_to_bf16u(embed[(size_t)sent[bt] * EE + e]);
}

// ---------------------------------------------------------------------------
// Input-projection GEMM:  Z[m][n] = sum_k A[m][k]*W[n][k] + bih[n] + bhh[n]
// A: bf16 (MROWS x K), W: bf16 (GG x K), Z: f32 (MROWS x GG)
// grid (MROWS/64, GG/64), block 256 (8 waves). Wave: 16 rows x 32 cols.
// ---------------------------------------------------------------------------
__global__ __launch_bounds__(256)
void gemm_in_proj(const unsigned short* __restrict__ A,
                  const unsigned short* __restrict__ W,
                  const float* __restrict__ bih, const float* __restrict__ bhh,
                  float* __restrict__ Z, int K) {
    int wave = threadIdx.x >> 5;
    int lane = threadIdx.x & 31;
    int m0 = blockIdx.x * 64 + (wave & 3) * 16;
    int n0 = blockIdx.y * 64 + (wave >> 2) * 32;

    v8f acc0 = {0,0,0,0,0,0,0,0};
    v8f acc1 = {0,0,0,0,0,0,0,0};

    int nkt = K >> 5;
    for (int kt = 0; kt < nkt; ++kt) {
        FragBF a, b0, b1;
        load_a_frag(a,  A, m0, kt * 32, K, lane);
        load_b_frag(b0, W, n0,      kt * 32, K, lane);
        load_b_frag(b1, W, n0 + 16, kt * 32, K, lane);
        acc0 = wmma_bf16(a, b0, acc0);
        acc1 = wmma_bf16(a, b1, acc1);
    }

    int laneN  = lane & 15;
    int rowoff = (lane >= 16) ? 8 : 0;
    int c0 = n0 + laneN;
    int c1 = n0 + 16 + laneN;
    float bia0 = bih[c0] + bhh[c0];
    float bia1 = bih[c1] + bhh[c1];
#pragma unroll
    for (int v = 0; v < 8; ++v) {
        int row = m0 + v + rowoff;
        Z[(size_t)row * GG + c0] = acc0[v] + bia0;
        Z[(size_t)row * GG + c1] = acc1[v] + bia1;
    }
}

// ---------------------------------------------------------------------------
// LSTM recurrence. Batch rows are independent, so split B=64 into 4 blocks of
// one 16-row WMMA tile per direction: grid = 8 blocks (blockIdx: dir*4+btile),
// 1024 threads = 32 waves. Wave w<16 owns gates (i,f) for hidden slice
// j=[16w,16w+16); wave w>=16 owns (g,o) for the same slice and publishes them
// via a 32KB LDS exchange. Whh fragments for a wave's two gate tiles
// (2 gates x 8 k-steps x 8 VGPR = 128 VGPRs) are PRELOADED into registers and
// stay resident for all 512 steps -> zero weight traffic on the serial path.
// h lives in LDS as bf16; cell state c lives in registers of waves 0-15.
// ---------------------------------------------------------------------------
#define HS 264   // padded LDS row stride (elements)

__global__ __launch_bounds__(1024)
void lstm_recur(const float* __restrict__ Zf, const float* __restrict__ Zb,
                const unsigned short* __restrict__ WhhF,
                const unsigned short* __restrict__ WhhB,
                unsigned short* __restrict__ out_bf,
                float* __restrict__ out_f, int f32_out) {
    __shared__ unsigned short hsh[16 * HS];      // 8.25 KB  (h, bf16)
    __shared__ float gex[2][16][HH];             // 32 KB    (g,o exchange)

    int dir   = blockIdx.x >> 2;
    int btile = blockIdx.x & 3;
    const float* Z = dir ? Zb : Zf;
    const unsigned short* Whh = dir ? WhhB : WhhF;
    int colofs = dir ? HH : 0;

    int wave = threadIdx.x >> 5;
    int lane = threadIdx.x & 31;
    int half = (wave >= 16);                 // 0: i,f   1: g,o
    int j0   = (wave & 15) * 16;
    int laneN  = lane & 15;
    int rowoff = (lane >= 16) ? 8 : 0;
    int row0 = btile * 16;                   // global batch-row base
    int g0 = half ? 2 : 0;                   // first owned gate

    // resident Whh fragments for this wave's two gate tiles
    FragBF breg[2][8];
#pragma unroll
    for (int gi = 0; gi < 2; ++gi)
#pragma unroll
        for (int kt = 0; kt < 8; ++kt)
            load_b_frag(breg[gi][kt], Whh, (g0 + gi) * HH + j0, kt * 32, HH, lane);

    for (int i = threadIdx.x; i < 16 * HS; i += 1024) hsh[i] = 0;
    __syncthreads();

    float cst[8];
#pragma unroll
    for (int v = 0; v < 8; ++v) cst[v] = 0.0f;

    for (int t = 0; t < TT; ++t) {
        int ta = dir ? (TT - 1 - t) : t;

        // accumulators start from the precomputed input projection (+biases)
        v8f acc[2];
#pragma unroll
        for (int gi = 0; gi < 2; ++gi) {
            int col = (g0 + gi) * HH + j0 + laneN;
            v8f a;
#pragma unroll
            for (int v = 0; v < 8; ++v) {
                int brow = row0 + v + rowoff;
                a[v] = Z[((size_t)brow * TT + ta) * GG + col];
            }
            acc[gi] = a;
        }
        // prefetch next step's Z slice into cache (global_prefetch_b8)
        if (t + 1 < TT) {
            int tn = dir ? (ta - 1) : (ta + 1);
            __builtin_prefetch(&Z[((size_t)(row0 + rowoff) * TT + tn) * GG + g0 * HH + j0 + laneN], 0, 0);
            __builtin_prefetch(&Z[((size_t)(row0 + rowoff) * TT + tn) * GG + (g0 + 1) * HH + j0 + laneN], 0, 0);
        }

        // gates += h @ Whh.T  (K=256, Whh in registers, h from LDS)
#pragma unroll
        for (int kt = 0; kt < 8; ++kt) {
            FragBF afr;
            load_a_frag(afr, hsh, 0, kt * 32, HS, lane);
            acc[0] = wmma_bf16(afr, breg[0][kt], acc[0]);
            acc[1] = wmma_bf16(afr, breg[1][kt], acc[1]);
        }

        if (half) {  // publish g,o tiles
#pragma unroll
            for (int gi = 0; gi < 2; ++gi)
#pragma unroll
                for (int v = 0; v < 8; ++v)
                    gex[gi][v + rowoff][j0 + laneN] = acc[gi][v];
        }
        __syncthreads();   // gex ready; everyone done reading old h

        if (!half) {       // register-local cell update (waves 0-15 own c)
#pragma unroll
            for (int v = 0; v < 8; ++v) {
                int r = v + rowoff;
                int j = j0 + laneN;
                float ig = sigf(acc[0][v]);
                float fg = sigf(acc[1][v]);
                float gg = tanhf(gex[0][r][j]);
                float og = sigf(gex[1][r][j]);
                float c  = fg * cst[v] + ig * gg;
                cst[v] = c;
                float h  = og * tanhf(c);
                hsh[r * HS + j] = f32_to_bf16u(h);
                size_t oidx = ((size_t)(row0 + r) * TT + ta) * DD + colofs + j;
                if (f32_out) out_f[oidx]  = h;
                else         out_bf[oidx] = f32_to_bf16u(h);
            }
        }
        __syncthreads();   // new h visible before next step
    }
}

// ---------------------------------------------------------------------------
// LayerNorm over last dim (512). One block (256 threads) per (b,t) row.
// Emits bf16 normed activations (input to the WMMA output projection).
// ---------------------------------------------------------------------------
__global__ __launch_bounds__(256)
void ln_kernel(const float* __restrict__ X, const float* __restrict__ gamma,
               const float* __restrict__ beta, unsigned short* __restrict__ Y) {
    __shared__ float red[256];
    int row = blockIdx.x;
    const float* x = X + (size_t)row * DD;
    float s = 0.f, s2 = 0.f;
    for (int i = threadIdx.x; i < DD; i += 256) { float v = x[i]; s += v; s2 += v * v; }
    red[threadIdx.x] = s; __syncthreads();
    for (int o = 128; o > 0; o >>= 1) { if (threadIdx.x < o) red[threadIdx.x] += red[threadIdx.x + o]; __syncthreads(); }
    float mu = red[0] * (1.0f / DD); __syncthreads();
    red[threadIdx.x] = s2; __syncthreads();
    for (int o = 128; o > 0; o >>= 1) { if (threadIdx.x < o) red[threadIdx.x] += red[threadIdx.x + o]; __syncthreads(); }
    float var = red[0] * (1.0f / DD) - mu * mu;
    float rs = rsqrtf(var + 1e-5f);
    for (int i = threadIdx.x; i < DD; i += 256)
        Y[(size_t)row * DD + i] = f32_to_bf16u((x[i] - mu) * rs * gamma[i] + beta[i]);
}

// ---------------------------------------------------------------------------
// Output projection via WMMA: feats[m][n] = normed[m] . Wout[n] + bout[n]
// N padded 35->48 (3 n-tiles). grid MROWS/64, block 128 (4 waves, 16 rows each).
// ---------------------------------------------------------------------------
__global__ __launch_bounds__(128)
void proj_wmma(const unsigned short* __restrict__ Xbf,
               const unsigned short* __restrict__ Wbf,   // 48x512 bf16 (padded)
               const float* __restrict__ bout, float* __restrict__ feats) {
    int wave = threadIdx.x >> 5;
    int lane = threadIdx.x & 31;
    int m0 = blockIdx.x * 64 + wave * 16;

    v8f acc[3];
#pragma unroll
    for (int nt = 0; nt < 3; ++nt) acc[nt] = (v8f){0,0,0,0,0,0,0,0};

    for (int kt = 0; kt < DD / 32; ++kt) {
        FragBF a;
        load_a_frag(a, Xbf, m0, kt * 32, DD, lane);
#pragma unroll
        for (int nt = 0; nt < 3; ++nt) {
            FragBF b;
            load_b_frag(b, Wbf, nt * 16, kt * 32, DD, lane);
            acc[nt] = wmma_bf16(a, b, acc[nt]);
        }
    }

    int laneN  = lane & 15;
    int rowoff = (lane >= 16) ? 8 : 0;
#pragma unroll
    for (int nt = 0; nt < 3; ++nt) {
        int c = nt * 16 + laneN;
        if (c < KK) {
            float bo = bout[c];
#pragma unroll
            for (int v = 0; v < 8; ++v) {
                int row = m0 + v + rowoff;
                feats[(size_t)row * KK + c] = acc[nt][v] + bo;
            }
        }
    }
}

// ---------------------------------------------------------------------------
// CRF forward (all-path logsumexp scan). One block (64 thr) per batch element.
// ---------------------------------------------------------------------------
__global__ __launch_bounds__(64)
void crf_all_kernel(const float* __restrict__ feats, const float* __restrict__ trans,
                    float* __restrict__ all_sc) {
    __shared__ float tr[KK * KK];
    __shared__ float alpha[40], nalpha[40];
    int b = blockIdx.x, tid = threadIdx.x;
    for (int i = tid; i < KK * KK; i += 64) tr[i] = trans[i];
    if (tid < KK) alpha[tid] = (tid == KSTART) ? 0.0f : -10000.0f;
    __syncthreads();
    for (int t = 0; t < TT; ++t) {
        if (tid < KK) {
            float emit = feats[((size_t)b * TT + t) * KK + tid];
            float m = -3.4e38f;
            for (int j = 0; j < KK; ++j) m = fmaxf(m, alpha[j] + tr[tid * KK + j]);
            float s = 0.f;
            for (int j = 0; j < KK; ++j) s += __expf(alpha[j] + tr[tid * KK + j] - m);
            nalpha[tid] = emit + m + __logf(s);
        }
        __syncthreads();
        if (tid < KK) alpha[tid] = nalpha[tid];
        __syncthreads();
    }
    if (tid == 0) {
        float m = -3.4e38f;
        for (int i = 0; i < KK; ++i) m = fmaxf(m, alpha[i] + tr[KSTOP * KK + i]);
        float s = 0.f;
        for (int i = 0; i < KK; ++i) s += __expf(alpha[i] + tr[KSTOP * KK + i] - m);
        all_sc[b] = m + __logf(s);
    }
}

__global__ __launch_bounds__(256)
void crf_real_kernel(const float* __restrict__ feats, const float* __restrict__ trans,
                     const int* __restrict__ tags, float* __restrict__ real_sc) {
    __shared__ float red[256];
    int b = blockIdx.x, tid = threadIdx.x;
    float s = 0.f;
    for (int t = tid; t < TT; t += 256) {
        int cur = tags[b * TT + t];
        int prev = (t == 0) ? KSTART : tags[b * TT + t - 1];
        s += trans[cur * KK + prev] + feats[((size_t)b * TT + t) * KK + cur];
    }
    red[tid] = s; __syncthreads();
    for (int o = 128; o > 0; o >>= 1) { if (tid < o) red[tid] += red[tid + o]; __syncthreads(); }
    if (tid == 0) real_sc[b] = red[0] + trans[KSTOP * KK + tags[b * TT + TT - 1]];
}

__global__ __launch_bounds__(64)
void final_kernel(const float* __restrict__ all_sc, const float* __restrict__ real_sc,
                  float* __restrict__ out) {
    __shared__ float red[64];
    int tid = threadIdx.x;
    red[tid] = all_sc[tid] - real_sc[tid];
    __syncthreads();
    for (int o = 32; o > 0; o >>= 1) { if (tid < o) red[tid] += red[tid + o]; __syncthreads(); }
    if (tid == 0) out[0] = red[0] * (1.0f / BB);
}

// ---------------------------------------------------------------------------
// Host launch
// ---------------------------------------------------------------------------
extern "C" void kernel_launch(void* const* d_in, const int* in_sizes, int n_in,
                              void* d_out, int out_size, void* d_ws, size_t ws_size,
                              hipStream_t stream) {
    const int*   sentence = (const int*)  d_in[0];
    const int*   tags     = (const int*)  d_in[2];
    const float* embed    = (const float*)d_in[3];
    const float* Wih_l0f = (const float*)d_in[4];
    const float* Whh_l0f = (const float*)d_in[5];
    const float* bih_l0f = (const float*)d_in[6];
    const float* bhh_l0f = (const float*)d_in[7];
    const float* Wih_l0b = (const float*)d_in[8];
    const float* Whh_l0b = (const float*)d_in[9];
    const float* bih_l0b = (const float*)d_in[10];
    const float* bhh_l0b = (const float*)d_in[11];
    const float* Wih_l1f = (const float*)d_in[12];
    const float* Whh_l1f = (const float*)d_in[13];
    const float* bih_l1f = (const float*)d_in[14];
    const float* bhh_l1f = (const float*)d_in[15];
    const float* Wih_l1b = (const float*)d_in[16];
    const float* Whh_l1b = (const float*)d_in[17];
    const float* bih_l1b = (const float*)d_in[18];
    const float* bhh_l1b = (const float*)d_in[19];
    const float* ln_gamma = (const float*)d_in[20];
    const float* ln_beta  = (const float*)d_in[21];
    const float* Wout     = (const float*)d_in[22];
    const float* bout     = (const float*)d_in[23];
    const float* trans    = (const float*)d_in[24];

    // workspace carve-up
    char* base = (char*)d_ws;
    size_t off = 0;
    auto carve = [&](size_t bytes) -> void* {
        void* p = base + off;
        off = (off + bytes + 255) & ~(size_t)255;
        return p;
    };
    unsigned short* wih0f = (unsigned short*)carve((size_t)GG * EE * 2);
    unsigned short* wih0b = (unsigned short*)carve((size_t)GG * EE * 2);
    unsigned short* wih1f = (unsigned short*)carve((size_t)GG * DD * 2);
    unsigned short* wih1b = (unsigned short*)carve((size_t)GG * DD * 2);
    unsigned short* whh0f = (unsigned short*)carve((size_t)GG * HH * 2);
    unsigned short* whh0b = (unsigned short*)carve((size_t)GG * HH * 2);
    unsigned short* whh1f = (unsigned short*)carve((size_t)GG * HH * 2);
    unsigned short* whh1b = (unsigned short*)carve((size_t)GG * HH * 2);
    unsigned short* wout_bf = (unsigned short*)carve((size_t)NPAD * DD * 2);
    unsigned short* x_bf  = (unsigned short*)carve((size_t)MROWS * EE * 2);
    unsigned short* x1_bf = (unsigned short*)carve((size_t)MROWS * DD * 2);  // later reused as normed bf16
    float* Zf    = (float*)carve((size_t)MROWS * GG * 4);   // reused layer0->layer1
    float* Zb    = (float*)carve((size_t)MROWS * GG * 4);
    float* outbf = (float*)carve((size_t)MROWS * DD * 4);
    float* feats = (float*)carve((size_t)MROWS * KK * 4);
    float* all_sc  = (float*)carve(BB * 4);
    float* real_sc = (float*)carve(BB * 4);

    auto cvt = [&](const float* s, unsigned short* d, int n) {
        cvt_f32_bf16_kernel<<<(n + 255) / 256, 256, 0, stream>>>(s, d, n);
    };
    cvt(Wih_l0f, wih0f, GG * EE);
    cvt(Wih_l0b, wih0b, GG * EE);
    cvt(Wih_l1f, wih1f, GG * DD);
    cvt(Wih_l1b, wih1b, GG * DD);
    cvt(Whh_l0f, whh0f, GG * HH);
    cvt(Whh_l0b, whh0b, GG * HH);
    cvt(Whh_l1f, whh1f, GG * HH);
    cvt(Whh_l1b, whh1b, GG * HH);
    pad_wout_kernel<<<(NPAD * DD + 255) / 256, 256, 0, stream>>>(Wout, wout_bf);

    embed_kernel<<<(MROWS * EE + 255) / 256, 256, 0, stream>>>(sentence, embed, x_bf);

    dim3 ggrid(MROWS / 64, GG / 64);
    // layer 0 input projections
    gemm_in_proj<<<ggrid, 256, 0, stream>>>(x_bf, wih0f, bih_l0f, bhh_l0f, Zf, EE);
    gemm_in_proj<<<ggrid, 256, 0, stream>>>(x_bf, wih0b, bih_l0b, bhh_l0b, Zb, EE);
    // layer 0 recurrence -> x1 (bf16, concat fwd|bwd); 8 blocks (2 dir x 4 batch tiles)
    lstm_recur<<<8, 1024, 0, stream>>>(Zf, Zb, whh0f, whh0b, x1_bf, (float*)nullptr, 0);
    // layer 1 input projections (reuse Z buffers)
    gemm_in_proj<<<ggrid, 256, 0, stream>>>(x1_bf, wih1f, bih_l1f, bhh_l1f, Zf, DD);
    gemm_in_proj<<<ggrid, 256, 0, stream>>>(x1_bf, wih1b, bih_l1b, bhh_l1b, Zb, DD);
    // layer 1 recurrence -> outbf (f32, concat fwd|bwd)
    lstm_recur<<<8, 1024, 0, stream>>>(Zf, Zb, whh1f, whh1b, (unsigned short*)nullptr, outbf, 1);

    // LayerNorm -> bf16 normed (reuses x1_bf), then WMMA output projection
    ln_kernel<<<MROWS, 256, 0, stream>>>(outbf, ln_gamma, ln_beta, x1_bf);
    proj_wmma<<<MROWS / 64, 128, 0, stream>>>(x1_bf, wout_bf, bout, feats);

    // CRF
    crf_all_kernel<<<BB, 64, 0, stream>>>(feats, trans, all_sc);
    crf_real_kernel<<<BB, 256, 0, stream>>>(feats, trans, tags, real_sc);
    final_kernel<<<1, 64, 0, stream>>>(all_sc, real_sc, (float*)d_out);
}
